// GINEWithContextPooling_90692529422478
// MI455X (gfx1250) — compile-verified, matrix operand order checked by reading, not run on hardware
//
#include <hip/hip_runtime.h>
#include <hip/hip_bf16.h>

#define NN 100000
#define NE 1600000
#define NM 4000
#define SLOPE 0.01f
#define BN_EPS 1e-5f

typedef __attribute__((ext_vector_type(16))) _Float16 v16h;
typedef __attribute__((ext_vector_type(8)))  _Float16 v8h;
typedef __attribute__((ext_vector_type(4)))  _Float16 v4h;
typedef __attribute__((ext_vector_type(8)))  float    v8f;

// ---------------- utility ----------------
__global__ void zero_f32_kernel(float* __restrict__ p, long n) {
  long i = blockIdx.x * (long)blockDim.x + threadIdx.x;
  if (i < n) p[i] = 0.f;
}

__global__ void leaky_f32_kernel(float* __restrict__ p, long n) {
  long i = blockIdx.x * (long)blockDim.x + threadIdx.x;
  if (i < n) { float v = p[i]; p[i] = v >= 0.f ? v : SLOPE * v; }
}

// -------- fused GINE edge stage: e = ea@W+b ; msg = relu(x[src]+e) ; atomic scatter to aggr[dst]
template<int FIN>
__global__ void edge_kernel(const float* __restrict__ ea, const float* __restrict__ ew,
                            const float* __restrict__ eb, const float* __restrict__ x,
                            const int* __restrict__ src, const int* __restrict__ dst,
                            float* __restrict__ aggr) {
  constexpr int TPE = FIN / 4;
  __shared__ float sW[12 * FIN];
  __shared__ float sB[FIN];
  for (int i = threadIdx.x; i < 12 * FIN; i += blockDim.x) sW[i] = ew[i];
  for (int i = threadIdx.x; i < FIN; i += blockDim.x) sB[i] = eb[i];
  __syncthreads();
  long t = blockIdx.x * (long)blockDim.x + threadIdx.x;
  long total = (long)NE * TPE;
  if (t >= total) return;
  int e  = (int)(t / TPE);
  int c0 = (int)(t % TPE) * 4;
  int s = src[e], d = dst[e];
  float a[12];
#pragma unroll
  for (int f = 0; f < 12; ++f) a[f] = ea[e * 12 + f];
  float m0 = sB[c0 + 0], m1 = sB[c0 + 1], m2 = sB[c0 + 2], m3 = sB[c0 + 3];
#pragma unroll
  for (int f = 0; f < 12; ++f) {
    const float* wr = &sW[f * FIN + c0];
    m0 = fmaf(a[f], wr[0], m0);
    m1 = fmaf(a[f], wr[1], m1);
    m2 = fmaf(a[f], wr[2], m2);
    m3 = fmaf(a[f], wr[3], m3);
  }
  const float4 xv = *reinterpret_cast<const float4*>(x + (long)s * FIN + c0);
  m0 = fmaxf(xv.x + m0, 0.f);
  m1 = fmaxf(xv.y + m1, 0.f);
  m2 = fmaxf(xv.z + m2, 0.f);
  m3 = fmaxf(xv.w + m3, 0.f);
  float* ap = aggr + (long)d * FIN + c0;
  atomicAdd(ap + 0, m0);
  atomicAdd(ap + 1, m1);
  atomicAdd(ap + 2, m2);
  atomicAdd(ap + 3, m3);
}

// -------- h_pre = (1+eps)*x + aggr, converted to f16 for WMMA
__global__ void prep_kernel(const float* __restrict__ x, const float* __restrict__ aggr,
                            const float* __restrict__ epsPtr, _Float16* __restrict__ out, long n) {
  long i = blockIdx.x * (long)blockDim.x + threadIdx.x;
  if (i >= n) return;
  float eps = epsPtr[0];
  out[i] = (_Float16)((1.0f + eps) * x[i] + aggr[i]);
}

// -------- WMMA GEMM: out[N,DOUT] = A16[N,K] @ W[K,DOUT] + bias  (f32 accumulate)
template<int K, int DOUT>
__global__ void gemm_wmma_kernel(const _Float16* __restrict__ A, const float* __restrict__ W,
                                 const float* __restrict__ bias, float* __restrict__ out,
                                 int rowTiles) {
  __shared__ _Float16 sW[DOUT * K];  // transposed: sW[n*K + k]
  for (int i = threadIdx.x; i < K * DOUT; i += blockDim.x) {
    int k = i % K, n = i / K;
    sW[n * K + k] = (_Float16)W[k * DOUT + n];
  }
  __syncthreads();
  const int wave = threadIdx.x >> 5;
  const int lane = threadIdx.x & 31;
  const int tile = blockIdx.x * (blockDim.x >> 5) + wave;
  if (tile >= rowTiles) return;
  constexpr int CT = DOUT / 16;
  v8f acc[CT];
#pragma unroll
  for (int c = 0; c < CT; ++c)
#pragma unroll
    for (int j = 0; j < 8; ++j) acc[c][j] = 0.f;

  const int m  = lane & 15;
  const int kb = (lane >> 4) << 3;  // 0 or 8: CDNA5 16-bit A-frag K base per half-wave
  const _Float16* Arow = A + ((long)tile * 16 + m) * K;
#pragma unroll
  for (int k0 = 0; k0 < K; k0 += 32) {
    v8h alo = *reinterpret_cast<const v8h*>(Arow + k0 + kb);
    v8h ahi = *reinterpret_cast<const v8h*>(Arow + k0 + kb + 16);
    v16h afrag;
#pragma unroll
    for (int i = 0; i < 8; ++i) { afrag[i] = alo[i]; afrag[8 + i] = ahi[i]; }
#pragma unroll
    for (int c = 0; c < CT; ++c) {
      const _Float16* Wc = &sW[(c * 16 + m) * K + k0 + kb];
      v8h blo = *reinterpret_cast<const v8h*>(Wc);
      v8h bhi = *reinterpret_cast<const v8h*>(Wc + 16);
      v16h bfrag;
#pragma unroll
      for (int i = 0; i < 8; ++i) { bfrag[i] = blo[i]; bfrag[8 + i] = bhi[i]; }
      acc[c] = __builtin_amdgcn_wmma_f32_16x16x32_f16(
          false, afrag, false, bfrag, (short)0, acc[c], false, false);
    }
  }
  // C/D layout: vgpr r -> row rbase+r, col lane&15
  const int rbase = (lane >> 4) * 8;
#pragma unroll
  for (int c = 0; c < CT; ++c) {
    int col = c * 16 + (lane & 15);
    float bv = bias[col];
#pragma unroll
    for (int r = 0; r < 8; ++r) {
      long row = (long)tile * 16 + rbase + r;
      out[row * DOUT + col] = acc[c][r] + bv;
    }
  }
}

// -------- per-column sum / sumsq partial reduction (for training-mode BN)
template<int C>
__global__ void stats_kernel(const float* __restrict__ x, float* __restrict__ stats, int nrows) {
  constexpr int RPB_T = 256 / C;  // concurrent rows per block
  __shared__ float ssum[256], ssq[256];
  const int c  = threadIdx.x % C;
  const int rg = threadIdx.x / C;
  const int ROWS = 512;
  int r0 = blockIdx.x * ROWS;
  int rend = r0 + ROWS; if (rend > nrows) rend = nrows;
  float s = 0.f, q = 0.f;
  for (int r = r0 + rg; r < rend; r += RPB_T) {
    float v = x[(long)r * C + c];
    s += v; q += v * v;
  }
  ssum[threadIdx.x] = s; ssq[threadIdx.x] = q;
  __syncthreads();
  if (rg == 0) {
#pragma unroll
    for (int g = 1; g < RPB_T; ++g) { s += ssum[g * C + c]; q += ssq[g * C + c]; }
    atomicAdd(&stats[c], s);
    atomicAdd(&stats[C + c], q);
  }
}

// -------- BN apply + LeakyReLU; writes f16 (pre-GEMM) or f32 (inter-layer)
template<int C, bool F16OUT>
__global__ void bn_apply_kernel(const float* __restrict__ x, const float* __restrict__ stats,
                                const float* __restrict__ g, const float* __restrict__ b,
                                float* __restrict__ of32, _Float16* __restrict__ of16, int nrows) {
  long i = blockIdx.x * (long)blockDim.x + threadIdx.x;
  long total = (long)nrows * C;
  if (i >= total) return;
  int c = (int)(i % C);
  float invN = 1.0f / (float)nrows;
  float mean = stats[c] * invN;
  float var  = stats[C + c] * invN - mean * mean;
  float inv  = rsqrtf(fmaxf(var, 0.f) + BN_EPS);
  float y = g[c] * (x[i] - mean) * inv + b[c];
  y = y >= 0.f ? y : SLOPE * y;
  if (F16OUT) of16[i] = (_Float16)y;
  else        of32[i] = y;
}

// -------- global_add_pool via atomics
__global__ void pool_kernel(const float* __restrict__ x, const int* __restrict__ batch,
                            float* __restrict__ pool) {
  long t = blockIdx.x * (long)blockDim.x + threadIdx.x;
  long total = (long)NN * 16;
  if (t >= total) return;
  int n = (int)(t / 16), c0 = (int)(t % 16) * 4;
  int mol = batch[n];
  const float4 v = *reinterpret_cast<const float4*>(x + (long)n * 64 + c0);
  float* pp = pool + (long)mol * 64 + c0;
  atomicAdd(pp + 0, v.x);
  atomicAdd(pp + 1, v.y);
  atomicAdd(pp + 2, v.z);
  atomicAdd(pp + 3, v.w);
}

// -------- concat [x || pool[batch]] -> f16 [N,128]
__global__ void concat_kernel(const float* __restrict__ x, const float* __restrict__ pool,
                              const int* __restrict__ batch, _Float16* __restrict__ out) {
  long t = blockIdx.x * (long)blockDim.x + threadIdx.x;
  long total = (long)NN * 32;
  if (t >= total) return;
  int n = (int)(t / 32), c0 = (int)(t % 32) * 4;
  float4 v;
  if (c0 < 64) v = *reinterpret_cast<const float4*>(x + (long)n * 64 + c0);
  else         v = *reinterpret_cast<const float4*>(pool + (long)batch[n] * 64 + (c0 - 64));
  v4h o; o[0] = (_Float16)v.x; o[1] = (_Float16)v.y; o[2] = (_Float16)v.z; o[3] = (_Float16)v.w;
  *reinterpret_cast<v4h*>(out + (long)n * 128 + c0) = o;
}

// -------- final: out[n] = leaky(bn(C[n,:])) . w3 + b3   (wave-per-node matvec)
__global__ void final_kernel(const float* __restrict__ xin, const float* __restrict__ stats,
                             const float* __restrict__ g, const float* __restrict__ b,
                             const float* __restrict__ w3, const float* __restrict__ b3,
                             float* __restrict__ out, int nrows) {
  const int wave = threadIdx.x >> 5, lane = threadIdx.x & 31;
  const int n = blockIdx.x * (blockDim.x >> 5) + wave;
  if (n >= nrows) return;
  float acc = 0.f;
  float invN = 1.0f / (float)nrows;
#pragma unroll
  for (int j = 0; j < 4; ++j) {
    int c = lane + 32 * j;
    float mean = stats[c] * invN;
    float var  = stats[128 + c] * invN - mean * mean;
    float inv  = rsqrtf(fmaxf(var, 0.f) + BN_EPS);
    float y = g[c] * (xin[(long)n * 128 + c] - mean) * inv + b[c];
    y = y >= 0.f ? y : SLOPE * y;
    acc = fmaf(y, w3[c], acc);
  }
#pragma unroll
  for (int off = 16; off > 0; off >>= 1) acc += __shfl_down(acc, off, 32);
  if (lane == 0) out[n] = acc + b3[0];
}

// =================== host ===================
static inline int cdiv_l(long a, long b) { return (int)((a + b - 1) / b); }

extern "C" void kernel_launch(void* const* d_in, const int* in_sizes, int n_in,
                              void* d_out, int out_size, void* d_ws, size_t ws_size,
                              hipStream_t stream) {
  (void)in_sizes; (void)n_in; (void)out_size; (void)ws_size;
  // setup_inputs() insertion-order leaves:
  // 0:x 1:edge_attr 2..28:convs[3]x{eps,edge_w,edge_b,w1,b1,bn_g,bn_b,w2,b2}
  // 29..34:bns[3]x{g,b} 35..44:clf{w1,b1,g1,bb1,w2,b2,g2,bb2,w3,b3} 45:edge_index 46:batch
  const float* x0 = (const float*)d_in[0];
  const float* ea = (const float*)d_in[1];
  auto cv = [&](int i, int f) { return (const float*)d_in[2 + 9 * i + f]; };
  const int* eidx  = (const int*)d_in[45];
  const int* batch = (const int*)d_in[46];
  const int* src = eidx;
  const int* dst = eidx + NE;

  // workspace carve-up (256B aligned)
  char* w = (char*)d_ws;
  auto carve = [&](size_t bytes) { char* p = w; w += (bytes + 255) & ~(size_t)255; return (void*)p; };
  float*    A    = (float*)   carve((size_t)NN * 64 * 4);   // node features
  float*    Bb   = (float*)   carve((size_t)NN * 64 * 4);   // aggr / hidden
  float*    Cc   = (float*)   carve((size_t)NN * 128 * 4);  // clf hidden
  _Float16* H16  = (_Float16*)carve((size_t)NN * 128 * 2);  // f16 GEMM input staging
  float*    pool = (float*)   carve((size_t)NM * 64 * 4);
  float*    st   = (float*)   carve(256 * 4);               // [sum | sumsq]

  const int T = 256;
  const int rowTiles = NN / 16;                 // 6250
  const int gemmBlocks = (rowTiles + 7) / 8;    // 8 waves / block
  const int statBlocks = (NN + 511) / 512;

  for (int i = 0; i < 3; ++i) {
    const bool first = (i == 0);
    const long finN = first ? (long)NN * 32 : (long)NN * 64;
    const float* xin = first ? x0 : A;

    zero_f32_kernel<<<cdiv_l(finN, T), T, 0, stream>>>(Bb, finN);
    if (first)
      edge_kernel<32><<<cdiv_l((long)NE * 8, T), T, 0, stream>>>(ea, cv(i,1), cv(i,2), xin, src, dst, Bb);
    else
      edge_kernel<64><<<cdiv_l((long)NE * 16, T), T, 0, stream>>>(ea, cv(i,1), cv(i,2), xin, src, dst, Bb);

    prep_kernel<<<cdiv_l(finN, T), T, 0, stream>>>(xin, Bb, cv(i,0), H16, finN);

    if (first)
      gemm_wmma_kernel<32, 64><<<gemmBlocks, T, 0, stream>>>(H16, cv(i,3), cv(i,4), Bb, rowTiles);
    else
      gemm_wmma_kernel<64, 64><<<gemmBlocks, T, 0, stream>>>(H16, cv(i,3), cv(i,4), Bb, rowTiles);

    zero_f32_kernel<<<1, 128, 0, stream>>>(st, 128);
    stats_kernel<64><<<statBlocks, T, 0, stream>>>(Bb, st, NN);
    bn_apply_kernel<64, true><<<cdiv_l((long)NN * 64, T), T, 0, stream>>>(
        Bb, st, cv(i,5), cv(i,6), nullptr, H16, NN);

    gemm_wmma_kernel<64, 64><<<gemmBlocks, T, 0, stream>>>(H16, cv(i,7), cv(i,8), A, rowTiles);

    if (i < 2) {
      zero_f32_kernel<<<1, 128, 0, stream>>>(st, 128);
      stats_kernel<64><<<statBlocks, T, 0, stream>>>(A, st, NN);
      bn_apply_kernel<64, false><<<cdiv_l((long)NN * 64, T), T, 0, stream>>>(
          A, st, (const float*)d_in[29 + 2 * i], (const float*)d_in[30 + 2 * i], A, nullptr, NN);
    } else {
      leaky_f32_kernel<<<cdiv_l((long)NN * 64, T), T, 0, stream>>>(A, (long)NN * 64);
    }
  }

  // pooling + context concat
  zero_f32_kernel<<<cdiv_l((long)NM * 64, T), T, 0, stream>>>(pool, (long)NM * 64);
  pool_kernel<<<cdiv_l((long)NN * 16, T), T, 0, stream>>>(A, batch, pool);
  concat_kernel<<<cdiv_l((long)NN * 32, T), T, 0, stream>>>(A, pool, batch, H16);

  // classifier
  gemm_wmma_kernel<128, 128><<<gemmBlocks, T, 0, stream>>>(
      H16, (const float*)d_in[35], (const float*)d_in[36], Cc, rowTiles);
  zero_f32_kernel<<<1, 256, 0, stream>>>(st, 256);
  stats_kernel<128><<<statBlocks, T, 0, stream>>>(Cc, st, NN);
  bn_apply_kernel<128, true><<<cdiv_l((long)NN * 128, T), T, 0, stream>>>(
      Cc, st, (const float*)d_in[37], (const float*)d_in[38], nullptr, H16, NN);

  gemm_wmma_kernel<128, 128><<<gemmBlocks, T, 0, stream>>>(
      H16, (const float*)d_in[39], (const float*)d_in[40], Cc, rowTiles);
  zero_f32_kernel<<<1, 256, 0, stream>>>(st, 256);
  stats_kernel<128><<<statBlocks, T, 0, stream>>>(Cc, st, NN);

  final_kernel<<<(NN + 7) / 8, T, 0, stream>>>(
      Cc, st, (const float*)d_in[41], (const float*)d_in[42],
      (const float*)d_in[43], (const float*)d_in[44], (float*)d_out, NN);
}
